// MHSA_CGLU_755914244675
// MI455X (gfx1250) — compile-verified, matrix operand order checked by reading, not run on hardware
//
#include <hip/hip_runtime.h>
#include <hip/hip_bf16.h>

// Problem constants (from reference): B=8, C=256, H=W=32, N=1024,
// h_qkv=512, NH=8, KD=16, HD=32, hid=170, 2*hid=340.
#define BB   8
#define CC   256
#define HH   32
#define WWD  32
#define NN   1024
#define HQ   512
#define HID  170
#define FC1R 340
#define LEPS 1e-5f

typedef __attribute__((ext_vector_type(16))) __bf16 v16bf;
typedef __attribute__((ext_vector_type(8)))  float  v8f;

static __device__ inline v8f wmma_bf16(v16bf a, v16bf b, v8f c) {
  // D(16x16,f32) = A(16x32,bf16) x B(32x16,bf16) + C
  return __builtin_amdgcn_wmma_f32_16x16x32_bf16(false, a, false, b, (short)0, c,
                                                 false, false);
}

// Build one bf16 WMMA operand (8 VGPRs / lane = 16 bf16 values).
// Layout (per CDNA5 ISA 16-bit A matrix, B mirrored with N across lanes):
//   lane r = lane&15 is the M index (A) or N index (B); h = lane>>4 selects K+8.
//   VGPR v packs K pair: K = (v<4 ? 2v : 16+2(v-4)) + 8h.
// For A operand pass f(m,k) -> A[m][k]. For B operand pass f(n,k) -> B[k][n].
template <typename F>
static __device__ inline v16bf fill_tile(F f) {
  const int lane = threadIdx.x & 31;
  const int r = lane & 15;
  const int h = lane >> 4;
  v16bf t;
#pragma unroll
  for (int v = 0; v < 8; ++v) {
    const int kk = ((v < 4) ? (v << 1) : (16 + ((v - 4) << 1))) + (h << 3);
    t[2 * v]     = (__bf16)f(r, kk);
    t[2 * v + 1] = (__bf16)f(r, kk + 1);
  }
  return t;
}

// K-looped 16x16 output tile: fa(m,k)=A[m][k], fb(k,n)=B[k][n].
template <typename FA, typename FB>
static __device__ inline v8f wmma_gemm_loop(int K, FA fa, FB fb) {
  v8f acc = {};
  for (int k0 = 0; k0 < K; k0 += 32) {
    v16bf A  = fill_tile([&](int m, int k) { return fa(m, k0 + k); });
    v16bf Bm = fill_tile([&](int n, int k) { return fb(k0 + k, n); });
    acc = wmma_bf16(A, Bm, acc);
  }
  return acc;
}

static __device__ inline float redmax16(float v) {
  v = fmaxf(v, __shfl_xor(v, 1));
  v = fmaxf(v, __shfl_xor(v, 2));
  v = fmaxf(v, __shfl_xor(v, 4));
  v = fmaxf(v, __shfl_xor(v, 8));
  return v;  // reduced within each 16-lane half
}
static __device__ inline float redsum16(float v) {
  v += __shfl_xor(v, 1);
  v += __shfl_xor(v, 2);
  v += __shfl_xor(v, 4);
  v += __shfl_xor(v, 8);
  return v;
}

// ---------------- K1/K6: LayerNorm over channels (per pixel) ----------------
__global__ __launch_bounds__(256) void ln2d_kernel(const float* __restrict__ x,
                                                   const float* __restrict__ g,
                                                   const float* __restrict__ be,
                                                   __bf16* __restrict__ out_bf,
                                                   float* __restrict__ out_f) {
  const int t = blockIdx.x * blockDim.x + threadIdx.x;  // BB*NN threads
  const int b = t >> 10, n = t & 1023;
  const float* base = x + (size_t)b * CC * NN + n;
  float s = 0.f, s2 = 0.f;
  for (int c = 0; c < CC; ++c) {
    const float v = base[(size_t)c * NN];
    s += v; s2 += v * v;
  }
  const float mu  = s * (1.f / CC);
  const float var = s2 * (1.f / CC) - mu * mu;
  const float inv = rsqrtf(var + LEPS);
  for (int c = 0; c < CC; ++c) {
    const float y = (base[(size_t)c * NN] - mu) * inv * g[c] + be[c];
    const size_t idx = (size_t)b * CC * NN + (size_t)c * NN + n;
    out_bf[idx] = (__bf16)y;
    if (out_f) out_f[idx] = y;
  }
}

// ---------------- K2: qkv = BN(qkv_w @ xn); write bf16 qkv + f32 v ----------
__global__ __launch_bounds__(256) void qkv_kernel(const __bf16* __restrict__ xn,
                                                  const float* __restrict__ w,
                                                  const float* __restrict__ bn,
                                                  __bf16* __restrict__ qkv,
                                                  float* __restrict__ vf) {
  const int wave = threadIdx.x >> 5;
  const int gid  = blockIdx.x * 8 + wave;            // 8*32*64 waves
  const int b  = gid >> 11;
  const int r  = gid & 2047;
  const int o0 = (r >> 6) << 4;
  const int n0 = (r & 63) << 4;
  const __bf16* xb = xn + (size_t)b * CC * NN;
  v8f acc = wmma_gemm_loop(
      CC,
      [&](int m, int k) { return w[(size_t)(o0 + m) * CC + k]; },
      [&](int k, int n) { return (float)xb[(size_t)k * NN + n0 + n]; });
  const int lane = threadIdx.x & 31, col = lane & 15, half = lane >> 4;
#pragma unroll
  for (int e = 0; e < 8; ++e) {
    const int o = o0 + e + 8 * half;
    const float sc = bn[o] * rsqrtf(bn[3 * HQ + o] + LEPS);
    const float y  = (acc[e] - bn[2 * HQ + o]) * sc + bn[HQ + o];
    qkv[(size_t)b * HQ * NN + (size_t)o * NN + n0 + col] = (__bf16)y;
    const int om = o & 63;
    if (om >= 32) {  // v rows -> f32 copy for the positional-encoding dwconv
      const int ch = (o >> 6) * 32 + (om - 32);
      vf[(size_t)b * CC * NN + (size_t)ch * NN + n0 + col] = y;
    }
  }
}

// ---------------- K3: flash attention, one wave = 16 query rows -------------
__global__ __launch_bounds__(256) void attn_kernel(const __bf16* __restrict__ qkv,
                                                   float* __restrict__ of) {
  __shared__ __bf16 Pt[8][32][16];  // per-wave P^T (m,n) staging
  __shared__ float  st[8][16];      // per-wave per-n stats (alpha / rowsum)
  const int wave = threadIdx.x >> 5, lane = threadIdx.x & 31;
  const int col = lane & 15, half = lane >> 4;
  const int bh = blockIdx.x >> 3;
  const int b = bh >> 3, hh = bh & 7;
  const int n0 = (((blockIdx.x & 7) << 3) + wave) << 4;
  const __bf16* qb = qkv + (size_t)b * HQ * NN + (size_t)(hh * 64) * NN;
  const __bf16* kb = qb + (size_t)16 * NN;
  const __bf16* vb = qb + (size_t)32 * NN;

  // A = q^T strip (16 n-rows x KD, K padded to 32 with zeros)
  v16bf Aq = fill_tile([&](int m, int k) {
    return (k < 16) ? (float)qb[(size_t)k * NN + n0 + m] : 0.f;
  });

  v8f O0 = {}, O1 = {};
  float rmax[8], rsum[8];
#pragma unroll
  for (int e = 0; e < 8; ++e) { rmax[e] = -3.0e38f; rsum[e] = 0.f; }

  for (int m0 = 0; m0 < NN; m0 += 32) {
    v16bf B0 = fill_tile([&](int n, int k) {
      return (k < 16) ? (float)kb[(size_t)k * NN + m0 + n] : 0.f;
    });
    v16bf B1 = fill_tile([&](int n, int k) {
      return (k < 16) ? (float)kb[(size_t)k * NN + m0 + 16 + n] : 0.f;
    });
    v8f Z = {};
    v8f S0 = wmma_bf16(Aq, B0, Z);
    v8f S1 = wmma_bf16(Aq, B1, Z);

    float P0[8], P1[8], alf[8];
#pragma unroll
    for (int e = 0; e < 8; ++e) {
      const float s0 = S0[e] * 0.25f, s1 = S1[e] * 0.25f;  // KD^-0.5
      const float cm = redmax16(fmaxf(s0, s1));
      const float nm = fmaxf(rmax[e], cm);
      alf[e] = __expf(rmax[e] - nm);
      rmax[e] = nm;
      P0[e] = __expf(s0 - nm);
      P1[e] = __expf(s1 - nm);
      rsum[e] = rsum[e] * alf[e] + redsum16(P0[e] + P1[e]);
    }
    __syncthreads();  // previous iteration's Pt/st reads complete
    if (col == 0) {
#pragma unroll
      for (int e = 0; e < 8; ++e) st[wave][e + 8 * half] = alf[e];
    }
#pragma unroll
    for (int e = 0; e < 8; ++e) {      // S tile layout: row M = n, col N = m
      Pt[wave][col][e + 8 * half]      = (__bf16)P0[e];
      Pt[wave][16 + col][e + 8 * half] = (__bf16)P1[e];
    }
    __syncthreads();

    const float a_n = st[wave][col];   // O tile: col index = n
#pragma unroll
    for (int e = 0; e < 8; ++e) { O0[e] *= a_n; O1[e] *= a_n; }

    v16bf Bp = fill_tile([&](int n, int k) { return (float)Pt[wave][k][n]; });
    v16bf Av0 = fill_tile([&](int d, int k) {
      return (float)vb[(size_t)d * NN + m0 + k];
    });
    v16bf Av1 = fill_tile([&](int d, int k) {
      return (float)vb[(size_t)(16 + d) * NN + m0 + k];
    });
    O0 = wmma_bf16(Av0, Bp, O0);
    O1 = wmma_bf16(Av1, Bp, O1);
  }

  __syncthreads();
  if (col == 0) {
#pragma unroll
    for (int e = 0; e < 8; ++e) st[wave][e + 8 * half] = rsum[e];
  }
  __syncthreads();
  const float inv = 1.f / st[wave][col];
#pragma unroll
  for (int e = 0; e < 8; ++e) {
    const int d = e + 8 * half;
    of[(size_t)b * CC * NN + (size_t)(hh * 32 + d) * NN + n0 + col] = O0[e] * inv;
    of[(size_t)b * CC * NN + (size_t)(hh * 32 + 16 + d) * NN + n0 + col] = O1[e] * inv;
  }
}

// ---------------- K4: o += BN(dwconv3x3(v)); emit bf16 for proj GEMM --------
__global__ __launch_bounds__(256) void pe_kernel(const float* __restrict__ vf,
                                                 const float* __restrict__ w,
                                                 const float* __restrict__ bn,
                                                 const float* __restrict__ of,
                                                 __bf16* __restrict__ oo) {
  const int t = blockIdx.x * blockDim.x + threadIdx.x;  // BB*CC*NN
  const int n = t & 1023, c = (t >> 10) & 255, b = t >> 18;
  const int y = n >> 5, xw = n & 31;
  const float* vp = vf + (size_t)b * CC * NN + (size_t)c * NN;
  const float* wp = w + c * 9;
  float acc = 0.f;
#pragma unroll
  for (int i = 0; i < 3; ++i) {
    const int yy = y + i - 1;
    if (yy < 0 || yy > 31) continue;
#pragma unroll
    for (int j = 0; j < 3; ++j) {
      const int xx = xw + j - 1;
      if (xx < 0 || xx > 31) continue;
      acc += vp[yy * 32 + xx] * wp[i * 3 + j];
    }
  }
  const float sc = bn[c] * rsqrtf(bn[3 * CC + c] + LEPS);
  oo[t] = (__bf16)((acc - bn[2 * CC + c]) * sc + bn[CC + c] + of[t]);
}

// ---------------- K5: x1 = x + BN(proj_w @ oo) ------------------------------
__global__ __launch_bounds__(256) void proj_kernel(const __bf16* __restrict__ oo,
                                                   const float* __restrict__ w,
                                                   const float* __restrict__ bn,
                                                   const float* __restrict__ xin,
                                                   float* __restrict__ x1) {
  const int gid = blockIdx.x * 8 + (threadIdx.x >> 5);  // 8*16*64 waves
  const int b = gid >> 10, r = gid & 1023;
  const int o0 = (r >> 6) << 4, n0 = (r & 63) << 4;
  const __bf16* ob = oo + (size_t)b * CC * NN;
  v8f acc = wmma_gemm_loop(
      CC,
      [&](int m, int k) { return w[(size_t)(o0 + m) * CC + k]; },
      [&](int k, int n) { return (float)ob[(size_t)k * NN + n0 + n]; });
  const int lane = threadIdx.x & 31, col = lane & 15, half = lane >> 4;
#pragma unroll
  for (int e = 0; e < 8; ++e) {
    const int o = o0 + e + 8 * half;
    const float sc = bn[o] * rsqrtf(bn[3 * CC + o] + LEPS);
    const size_t idx = (size_t)b * CC * NN + (size_t)o * NN + n0 + col;
    x1[idx] = (acc[e] - bn[2 * CC + o]) * sc + bn[CC + o] + xin[idx];
  }
}

// ---------------- K7: h = fc1_w @ xn2 + fc1_b (rows padded to 352) ----------
__global__ __launch_bounds__(256) void fc1_kernel(const __bf16* __restrict__ xn2,
                                                  const float* __restrict__ w,
                                                  const float* __restrict__ bi,
                                                  float* __restrict__ h) {
  const int gid = blockIdx.x * 8 + (threadIdx.x >> 5);  // 8*22*64 waves
  const int b = gid / 1408, r = gid % 1408;
  const int o0 = (r >> 6) << 4, n0 = (r & 63) << 4;
  const __bf16* xb = xn2 + (size_t)b * CC * NN;
  v8f acc = wmma_gemm_loop(
      CC,
      [&](int m, int k) {
        return (o0 + m < FC1R) ? w[(size_t)(o0 + m) * CC + k] : 0.f;
      },
      [&](int k, int n) { return (float)xb[(size_t)k * NN + n0 + n]; });
  const int lane = threadIdx.x & 31, col = lane & 15, half = lane >> 4;
#pragma unroll
  for (int e = 0; e < 8; ++e) {
    const int o = o0 + e + 8 * half;
    if (o < FC1R)
      h[((size_t)b * FC1R + o) * NN + n0 + col] = acc[e] + bi[o];
  }
}

// ---------------- K8: act = gelu(dwconv3x3(a)+b) * g (bf16, K for fc2) ------
__global__ __launch_bounds__(256) void glu_kernel(const float* __restrict__ h,
                                                  const float* __restrict__ w,
                                                  const float* __restrict__ bi,
                                                  __bf16* __restrict__ act) {
  const int t = blockIdx.x * blockDim.x + threadIdx.x;  // BB*HID*NN
  const int n = t & 1023;
  const int rest = t >> 10;
  const int c = rest % HID, b = rest / HID;
  const int y = n >> 5, xw = n & 31;
  const float* ap = h + ((size_t)b * FC1R + c) * NN;
  const float* wp = w + c * 9;
  float acc = 0.f;
#pragma unroll
  for (int i = 0; i < 3; ++i) {
    const int yy = y + i - 1;
    if (yy < 0 || yy > 31) continue;
#pragma unroll
    for (int j = 0; j < 3; ++j) {
      const int xx = xw + j - 1;
      if (xx < 0 || xx > 31) continue;
      acc += ap[yy * 32 + xx] * wp[i * 3 + j];
    }
  }
  const float u  = acc + bi[c];
  const float ge = 0.5f * u * (1.f + erff(u * 0.70710678118f));  // exact gelu
  const float gv = h[((size_t)b * FC1R + HID + c) * NN + n];
  act[((size_t)b * HID + c) * NN + n] = (__bf16)(ge * gv);
}

// ---------------- K9: out = x1 + xn2 + fc2_w @ act + fc2_b (K pad 192) ------
__global__ __launch_bounds__(256) void fc2_kernel(const __bf16* __restrict__ act,
                                                  const float* __restrict__ w,
                                                  const float* __restrict__ bi,
                                                  const float* __restrict__ xn2f,
                                                  const float* __restrict__ x1,
                                                  float* __restrict__ out) {
  const int gid = blockIdx.x * 8 + (threadIdx.x >> 5);  // 8*16*64 waves
  const int b = gid >> 10, r = gid & 1023;
  const int o0 = (r >> 6) << 4, n0 = (r & 63) << 4;
  const __bf16* ab = act + (size_t)b * HID * NN;
  v8f acc = wmma_gemm_loop(
      192,
      [&](int m, int k) {
        return (k < HID) ? w[(size_t)(o0 + m) * HID + k] : 0.f;
      },
      [&](int k, int n) {
        return (k < HID) ? (float)ab[(size_t)k * NN + n0 + n] : 0.f;
      });
  const int lane = threadIdx.x & 31, col = lane & 15, half = lane >> 4;
#pragma unroll
  for (int e = 0; e < 8; ++e) {
    const int o = o0 + e + 8 * half;
    const size_t idx = (size_t)b * CC * NN + (size_t)o * NN + n0 + col;
    out[idx] = acc[e] + bi[o] + xn2f[idx] + x1[idx];
  }
}

extern "C" void kernel_launch(void* const* d_in, const int* in_sizes, int n_in,
                              void* d_out, int out_size, void* d_ws, size_t ws_size,
                              hipStream_t stream) {
  const float* x      = (const float*)d_in[0];
  const float* ln1_g  = (const float*)d_in[1];
  const float* ln1_b  = (const float*)d_in[2];
  const float* ln2_g  = (const float*)d_in[3];
  const float* ln2_b  = (const float*)d_in[4];
  const float* qkv_w  = (const float*)d_in[5];
  const float* qkv_bn = (const float*)d_in[6];
  const float* pe_w   = (const float*)d_in[7];
  const float* pe_bn  = (const float*)d_in[8];
  const float* proj_w = (const float*)d_in[9];
  const float* proj_bn= (const float*)d_in[10];
  const float* fc1_w  = (const float*)d_in[11];
  const float* fc1_b  = (const float*)d_in[12];
  const float* dw_w   = (const float*)d_in[13];
  const float* dw_b   = (const float*)d_in[14];
  const float* fc2_w  = (const float*)d_in[15];
  const float* fc2_b  = (const float*)d_in[16];
  float* out = (float*)d_out;

  char* ws = (char*)d_ws;
  size_t off = 0;
  auto take = [&](size_t bytes) -> void* {
    void* p = ws + off;
    off += (bytes + 255) & ~(size_t)255;
    return p;
  };
  const size_t BCN = (size_t)BB * CC * NN;
  __bf16* xn_bf  = (__bf16*)take(BCN * 2);
  __bf16* qkv_bf = (__bf16*)take((size_t)BB * HQ * NN * 2);
  float*  vf     = (float*) take(BCN * 4);
  float*  of     = (float*) take(BCN * 4);
  __bf16* oo_bf  = (__bf16*)take(BCN * 2);
  float*  x1     = (float*) take(BCN * 4);
  __bf16* xn2_bf = (__bf16*)take(BCN * 2);
  float*  xn2_f  = (float*) take(BCN * 4);
  float*  hbuf   = (float*) take((size_t)BB * FC1R * NN * 4);
  __bf16* act_bf = (__bf16*)take((size_t)BB * HID * NN * 2);
  (void)ws_size; (void)in_sizes; (void)n_in; (void)out_size;

  ln2d_kernel<<<(BB * NN) / 256, 256, 0, stream>>>(x, ln1_g, ln1_b, xn_bf, nullptr);
  qkv_kernel<<<2048, 256, 0, stream>>>(xn_bf, qkv_w, qkv_bn, qkv_bf, vf);
  attn_kernel<<<512, 256, 0, stream>>>(qkv_bf, of);
  pe_kernel<<<8192, 256, 0, stream>>>(vf, pe_w, pe_bn, of, oo_bf);
  proj_kernel<<<1024, 256, 0, stream>>>(oo_bf, proj_w, proj_bn, x, x1);
  ln2d_kernel<<<(BB * NN) / 256, 256, 0, stream>>>(x1, ln2_g, ln2_b, xn2_bf, xn2_f);
  fc1_kernel<<<1408, 256, 0, stream>>>(xn2_bf, fc1_w, fc1_b, hbuf);
  glu_kernel<<<(BB * HID * NN) / 256, 256, 0, stream>>>(hbuf, dw_w, dw_b, act_bf);
  fc2_kernel<<<1024, 256, 0, stream>>>(act_bf, fc2_w, fc2_b, xn2_f, x1, out);
}
